// MoDLayer_22488448761975
// MI455X (gfx1250) — compile-verified, matrix operand order checked by reading, not run on hardware
//
#include <hip/hip_runtime.h>
#include <cstdint>
#include <cstddef>

// ---------------- problem constants (match reference) ----------------
#define BB   4
#define TT   2048
#define DD   2048
#define HH   16
#define HDIM 128
#define II   5632
#define KK   512
#define SS   (BB*KK)          // 2048 packed tokens
#define EPSV 1e-6f

// ---------------- WMMA types / helpers ----------------
typedef __attribute__((ext_vector_type(16))) __bf16 v16bf;
typedef __attribute__((ext_vector_type(8)))  __bf16 v8bf;
typedef __attribute__((ext_vector_type(8)))  float  v8f;

// CDNA5 async global->LDS path (ASYNCcnt-tracked), with sync fallback.
#if defined(__gfx1250__)
#  if __has_builtin(__builtin_amdgcn_global_load_async_to_lds_b128)
#    define HAVE_ASYNC_LDS 1
#  endif
#endif
#ifndef HAVE_ASYNC_LDS
#  define HAVE_ASYNC_LDS 0
#endif

typedef int v4i __attribute__((vector_size(16)));
typedef __attribute__((address_space(1))) v4i as1_v4i;   // __device__ (global)
typedef __attribute__((address_space(3))) v4i as3_v4i;   // __shared__ (LDS)

__device__ __forceinline__ void async_ld_b128(const __bf16* g, __bf16* l) {
#if HAVE_ASYNC_LDS
  __builtin_amdgcn_global_load_async_to_lds_b128(
      (as1_v4i*)(__bf16*)g, (as3_v4i*)l, /*imm_offset=*/0, /*cpol=*/0);
#else
  *(v8bf*)l = *(const v8bf*)g;
#endif
}
__device__ __forceinline__ void async_wait0() {
#if HAVE_ASYNC_LDS
#  if __has_builtin(__builtin_amdgcn_s_wait_asynccnt)
  __builtin_amdgcn_s_wait_asynccnt(0);
#  else
  asm volatile("s_wait_asynccnt 0" ::: "memory");
#  endif
#endif
}

// Per-lane 16-bit WMMA operand fragment: two contiguous 16B chunks 16 elems
// apart (K {0..7,16..23} for lanes 0-15, {8..15,24..31} for lanes 16-31 when
// the caller offsets p by (lane>>4)*8).
__device__ __forceinline__ v16bf load_frag16(const __bf16* p) {
  v8bf lo = *(const v8bf*)(p);
  v8bf hi = *(const v8bf*)(p + 16);
  return __builtin_shufflevector(lo, hi, 0,1,2,3,4,5,6,7,8,9,10,11,12,13,14,15);
}

__device__ __forceinline__ v8f wmma_bf16(v16bf a, v16bf b, v8f c) {
  return __builtin_amdgcn_wmma_f32_16x16x32_bf16(
      /*neg_a=*/false, a, /*neg_b=*/false, b,
      /*c_mod=*/(short)0, c, /*reuse_a=*/false, /*reuse_b=*/false);
}

__device__ __forceinline__ float red32_sum(float v) {
#pragma unroll
  for (int off = 16; off; off >>= 1) v += __shfl_xor(v, off, 32);
  return v;
}
__device__ __forceinline__ float red16_sum(float v) {
#pragma unroll
  for (int off = 8; off; off >>= 1) v += __shfl_xor(v, off, 32);
  return v;
}
__device__ __forceinline__ float red16_max(float v) {
#pragma unroll
  for (int off = 8; off; off >>= 1) v = fmaxf(v, __shfl_xor(v, off, 32));
  return v;
}

// ---------------- 1) router scores: [B,T] = hidden . router_w + b ------------
__global__ void mod_router_kernel(const float* __restrict__ hs,
                                  const float* __restrict__ rw,
                                  const float* __restrict__ rb,
                                  float* __restrict__ scores) {
  const int token = blockIdx.x * 8 + (threadIdx.x >> 5);
  const int lane  = threadIdx.x & 31;
  const float* row = hs + (size_t)token * DD;
  float acc = 0.f;
  for (int i = lane; i < DD; i += 32) acc += row[i] * rw[i];
  acc = red32_sum(acc);
  if (lane == 0) scores[token] = acc + rb[0];
}

// ---------------- 2) per-row top-K via bitonic sort ----------------
__global__ void __launch_bounds__(1024)
mod_topk_kernel(const float* __restrict__ scores,
                int* __restrict__ idx_out, float* __restrict__ gate_out) {
  __shared__ float sv[TT];
  __shared__ int   si[TT];
  const int b = blockIdx.x, t = threadIdx.x;
  for (int i = t; i < TT; i += 1024) { sv[i] = scores[(size_t)b*TT + i]; si[i] = i; }
  __syncthreads();
  // sort descending by score (tie: smaller index first) — ascending under L
  for (int k = 2; k <= TT; k <<= 1) {
    for (int j = k >> 1; j > 0; j >>= 1) {
      for (int i = t; i < TT; i += 1024) {
        int ixj = i ^ j;
        if (ixj > i) {
          bool dir = ((i & k) == 0);
          float sa = sv[i], sb = sv[ixj];
          int   ia = si[i], ib = si[ixj];
          bool L = (sa > sb) || (sa == sb && ia < ib);
          if (L != dir) { sv[i] = sb; sv[ixj] = sa; si[i] = ib; si[ixj] = ia; }
        }
      }
      __syncthreads();
    }
  }
  // sort top-K entries ascending by original index (causal packed order)
  for (int k = 2; k <= KK; k <<= 1) {
    for (int j = k >> 1; j > 0; j >>= 1) {
      if (t < KK) {
        int i = t, ixj = i ^ j;
        if (ixj > i && ixj < KK) {
          bool dir = ((i & k) == 0);
          bool L = si[i] < si[ixj];
          if (L != dir) {
            int   ti = si[i]; si[i] = si[ixj]; si[ixj] = ti;
            float tf = sv[i]; sv[i] = sv[ixj]; sv[ixj] = tf;
          }
        }
      }
      __syncthreads();
    }
  }
  if (t < KK) {
    idx_out[(size_t)b*KK + t]  = si[t];
    gate_out[(size_t)b*KK + t] = 1.f / (1.f + expf(-sv[t]));
  }
}

// ---------------- 3) weight fp32 [R][C] -> bf16 transposed [C][R] ------------
__global__ void mod_transpose_cast_kernel(const float* __restrict__ src,
                                          __bf16* __restrict__ dst,
                                          int R, int C) {
  __shared__ float tile[32][33];
  const int c0 = blockIdx.x * 32, r0 = blockIdx.y * 32;
  for (int rr = threadIdx.y; rr < 32; rr += 8)
    tile[rr][threadIdx.x] = src[(size_t)(r0 + rr) * C + c0 + threadIdx.x];
  __syncthreads();
  for (int rr = threadIdx.y; rr < 32; rr += 8)
    dst[(size_t)(c0 + rr) * R + r0 + threadIdx.x] = (__bf16)tile[threadIdx.x][rr];
}

// ---------------- 4) gather + RMSNorm (+ RoPE tables). idx==null -> plain RMS -
__global__ void mod_gather_rms_kernel(const float* __restrict__ src,
                                      const int* __restrict__ idx,
                                      const float* __restrict__ lnw,
                                      float* __restrict__ x_out,
                                      __bf16* __restrict__ h_out,
                                      float* __restrict__ cosb,
                                      float* __restrict__ sinb) {
  const int s = blockIdx.x;
  const float* row;
  int pos = 0;
  if (idx) { int b = s / KK; pos = idx[s]; row = src + ((size_t)b * TT + pos) * DD; }
  else     { row = src + (size_t)s * DD; }
  float vals[8];
  float ss = 0.f;
#pragma unroll
  for (int r = 0; r < 8; ++r) {
    int c = threadIdx.x + r * 256;
    float v = row[c]; vals[r] = v; ss += v * v;
  }
  ss = red32_sum(ss);
  __shared__ float red[8];
  const int wid = threadIdx.x >> 5, lane = threadIdx.x & 31;
  if (lane == 0) red[wid] = ss;
  __syncthreads();
  if (threadIdx.x == 0) { float tot = 0.f; for (int i = 0; i < 8; ++i) tot += red[i]; red[0] = tot; }
  __syncthreads();
  const float scale = rsqrtf(red[0] / (float)DD + EPSV);
#pragma unroll
  for (int r = 0; r < 8; ++r) {
    int c = threadIdx.x + r * 256;
    if (x_out) x_out[(size_t)s * DD + c] = vals[r];
    h_out[(size_t)s * DD + c] = (__bf16)(vals[r] * scale * lnw[c]);
  }
  if (cosb && threadIdx.x < 64) {
    int d = threadIdx.x;
    float inv = powf(10000.0f, -(float)(2 * d) / (float)HDIM);
    float f = (float)pos * inv;
    float cv = cosf(f), sv2 = sinf(f);
    cosb[(size_t)s * HDIM + d] = cv;  cosb[(size_t)s * HDIM + d + 64] = cv;
    sinb[(size_t)s * HDIM + d] = sv2; sinb[(size_t)s * HDIM + d + 64] = sv2;
  }
}

// ---------------- 5) WMMA GEMM: C[M,N](f32) = A[M,K]bf16 . Bt[N,K]bf16^T -----
// 256 threads = 8 waves, block tile 128x128, K-step 64, double-buffered LDS
// filled by CDNA5 async global->LDS copies (ASYNCcnt) overlapped with WMMA.
__global__ void __launch_bounds__(256)
mod_gemm_kernel(const __bf16* __restrict__ A, const __bf16* __restrict__ Bt,
                float* __restrict__ C, const float* __restrict__ bias,
                const float* __restrict__ resid, int M, int N, int Kd) {
  __shared__ __bf16 lds_a[2][128 * 64];   // 2 x 16KB
  __shared__ __bf16 lds_b[2][128 * 64];   // 2 x 16KB
  const int n0 = blockIdx.x * 128, m0 = blockIdx.y * 128;
  const int t = threadIdx.x;
  const int wid = t >> 5, lane = t & 31, l16 = lane & 15, lh = lane >> 4;
  const int wm = wid & 1, wn = wid >> 1;   // 2 x 4 wave grid: 64x32 per wave
  v8f zc = {};
  v8f acc[4][2];
#pragma unroll
  for (int i = 0; i < 4; ++i)
#pragma unroll
    for (int j = 0; j < 2; ++j) acc[i][j] = zc;

  // loader mapping: thread t owns row (t&127), K half (t>>7)*32 -> 4x b128 each
  const int lr  = t & 127;
  const int lk0 = (t >> 7) * 32;
  const __bf16* ag = A  + (size_t)(m0 + lr) * Kd + lk0;
  const __bf16* bg = Bt + (size_t)(n0 + lr) * Kd + lk0;
  __bf16* la0 = &lds_a[0][lr * 64 + lk0];
  __bf16* lb0 = &lds_b[0][lr * 64 + lk0];
  __bf16* la1 = &lds_a[1][lr * 64 + lk0];
  __bf16* lb1 = &lds_b[1][lr * 64 + lk0];

  // preload tile 0 into buffer 0
#pragma unroll
  for (int c = 0; c < 4; ++c) {
    async_ld_b128(ag + c * 8, la0 + c * 8);
    async_ld_b128(bg + c * 8, lb0 + c * 8);
  }
  async_wait0();
  __syncthreads();

  int cur = 0;
  for (int k0 = 0; k0 < Kd; k0 += 64) {
    // prefetch next K-slab into the other buffer (overlaps with WMMAs below)
    if (k0 + 64 < Kd) {
      const __bf16* agn = ag + k0 + 64;
      const __bf16* bgn = bg + k0 + 64;
      __bf16* lan = cur ? la0 : la1;
      __bf16* lbn = cur ? lb0 : lb1;
#pragma unroll
      for (int c = 0; c < 4; ++c) {
        async_ld_b128(agn + c * 8, lan + c * 8);
        async_ld_b128(bgn + c * 8, lbn + c * 8);
      }
    }
    const __bf16* la = cur ? &lds_a[1][0] : &lds_a[0][0];
    const __bf16* lb = cur ? &lds_b[1][0] : &lds_b[0][0];
#pragma unroll
    for (int kk = 0; kk < 2; ++kk) {
      v16bf af[4], bfr[2];
#pragma unroll
      for (int i = 0; i < 4; ++i)
        af[i] = load_frag16(la + (wm * 64 + i * 16 + l16) * 64 + kk * 32 + lh * 8);
#pragma unroll
      for (int j = 0; j < 2; ++j)
        bfr[j] = load_frag16(lb + (wn * 32 + j * 16 + l16) * 64 + kk * 32 + lh * 8);
#pragma unroll
      for (int i = 0; i < 4; ++i)
#pragma unroll
        for (int j = 0; j < 2; ++j) acc[i][j] = wmma_bf16(af[i], bfr[j], acc[i][j]);
    }
    async_wait0();
    __syncthreads();
    cur ^= 1;
  }

#pragma unroll
  for (int i = 0; i < 4; ++i)
#pragma unroll
    for (int j = 0; j < 2; ++j)
#pragma unroll
      for (int r = 0; r < 8; ++r) {
        int row = m0 + wm * 64 + i * 16 + lh * 8 + r;
        int col = n0 + wn * 32 + j * 16 + l16;
        float v = acc[i][j][r];
        if (bias)  v += bias[col];
        if (resid) v += resid[(size_t)row * N + col];
        C[(size_t)row * N + col] = v;
      }
}

// ---------------- 6) RoPE + layout cast for attention ----------------
// qf/kf/vf: [S, D] f32 -> qh/kh: [H][S][HD] bf16, vt: [H][HD][S] bf16
__global__ void mod_rope_cast_kernel(const float* __restrict__ qf,
                                     const float* __restrict__ kf,
                                     const float* __restrict__ vf,
                                     const float* __restrict__ cosb,
                                     const float* __restrict__ sinb,
                                     __bf16* __restrict__ qh,
                                     __bf16* __restrict__ kh,
                                     __bf16* __restrict__ vt) {
  const int h = blockIdx.x, s = blockIdx.y, d = threadIdx.x;
  const size_t base = (size_t)s * DD + (size_t)h * HDIM;
  const float c  = cosb[(size_t)s * HDIM + d];
  const float sn = sinb[(size_t)s * HDIM + d];
  const float qv = qf[base + d];
  const float kv = kf[base + d];
  const float qp = (d < 64) ? -qf[base + d + 64] : qf[base + d - 64];
  const float kp = (d < 64) ? -kf[base + d + 64] : kf[base + d - 64];
  qh[((size_t)h * SS + s) * HDIM + d] = (__bf16)(qv * c + qp * sn);
  kh[((size_t)h * SS + s) * HDIM + d] = (__bf16)(kv * c + kp * sn);
  vt[((size_t)h * HDIM + d) * SS + s] = (__bf16)vf[base + d];
}

// ---------------- 7) flash attention: 64 queries x 1 head per block ----------
// 128 threads = 4 waves; wave owns 16 query rows. Q fragments in registers;
// K / V^T B-fragments streamed from global; P restaged via LDS (C->A layout).
__global__ void __launch_bounds__(128)
mod_flash_kernel(const __bf16* __restrict__ qh, const __bf16* __restrict__ kh,
                 const __bf16* __restrict__ vt, __bf16* __restrict__ attn) {
  __shared__ __bf16 lds_p[64 * 64];
  const int h  = blockIdx.y;
  const int q0 = blockIdx.x * 64;
  const int w = threadIdx.x >> 5, lane = threadIdx.x & 31;
  const int l16 = lane & 15, lh = lane >> 4;
  const int rowbase = q0 + w * 16 + lh * 8;
  const float sc_scale = 0.08838834764831845f;   // 1/sqrt(HD)

  const __bf16* qb = qh + ((size_t)h * SS + q0 + w * 16 + l16) * HDIM;
  v16bf qfr[4];
#pragma unroll
  for (int kk = 0; kk < 4; ++kk) qfr[kk] = load_frag16(qb + kk * 32 + lh * 8);

  v8f zc = {};
  v8f o[8];
#pragma unroll
  for (int j = 0; j < 8; ++j) o[j] = zc;
  float m_r[8], l_r[8];
#pragma unroll
  for (int r = 0; r < 8; ++r) { m_r[r] = -3.0e38f; l_r[r] = 0.f; }

  for (int j0 = 0; j0 <= q0; j0 += 64) {
    // ---- S = Q . K^T for this 64-key tile ----
    v8f sc[4];
#pragma unroll
    for (int nt = 0; nt < 4; ++nt) sc[nt] = zc;
#pragma unroll
    for (int kk = 0; kk < 4; ++kk) {
#pragma unroll
      for (int nt = 0; nt < 4; ++nt) {
        v16bf b = load_frag16(kh + ((size_t)h * SS + j0 + 16 * nt + l16) * HDIM
                              + kk * 32 + lh * 8);
        sc[nt] = wmma_bf16(qfr[kk], b, sc[nt]);
      }
    }
    // ---- scale + causal mask + online softmax ----
    float mnew[8];
#pragma unroll
    for (int r = 0; r < 8; ++r) mnew[r] = -3.0e38f;
#pragma unroll
    for (int nt = 0; nt < 4; ++nt) {
      int col = j0 + 16 * nt + l16;
#pragma unroll
      for (int r = 0; r < 8; ++r) {
        float v = sc[nt][r] * sc_scale;
        if (col > rowbase + r) v = -3.0e38f;
        sc[nt][r] = v;
        mnew[r] = fmaxf(mnew[r], v);
      }
    }
#pragma unroll
    for (int r = 0; r < 8; ++r) mnew[r] = red16_max(mnew[r]);
    float rs[8];
#pragma unroll
    for (int r = 0; r < 8; ++r) {
      float mc = fmaxf(m_r[r], mnew[r]);
      float corr = expf(m_r[r] - mc);
      m_r[r] = mc; l_r[r] *= corr;
#pragma unroll
      for (int j = 0; j < 8; ++j) o[j][r] *= corr;
      rs[r] = 0.f;
    }
#pragma unroll
    for (int nt = 0; nt < 4; ++nt)
#pragma unroll
      for (int r = 0; r < 8; ++r) {
        float p = expf(sc[nt][r] - m_r[r]);
        rs[r] += p;
        lds_p[(w * 16 + lh * 8 + r) * 64 + 16 * nt + l16] = (__bf16)p;
      }
#pragma unroll
    for (int r = 0; r < 8; ++r) l_r[r] += red16_sum(rs[r]);
    __syncthreads();
    // ---- O += P . V ----
#pragma unroll
    for (int kk2 = 0; kk2 < 2; ++kk2) {
      v16bf a = load_frag16(&lds_p[(w * 16 + l16) * 64 + kk2 * 32 + lh * 8]);
#pragma unroll
      for (int j = 0; j < 8; ++j) {
        v16bf b = load_frag16(vt + ((size_t)h * HDIM + 16 * j + l16) * SS
                              + j0 + kk2 * 32 + lh * 8);
        o[j] = wmma_bf16(a, b, o[j]);
      }
    }
    __syncthreads();
  }
  // ---- normalize + write attn [S][D] bf16 ----
#pragma unroll
  for (int j = 0; j < 8; ++j)
#pragma unroll
    for (int r = 0; r < 8; ++r) {
      int row = rowbase + r;
      attn[(size_t)row * DD + (size_t)h * HDIM + 16 * j + l16] =
          (__bf16)(o[j][r] / l_r[r]);
    }
}

// ---------------- 8) SwiGLU elementwise ----------------
__global__ void mod_silu_mul_kernel(const float* __restrict__ g,
                                    const float* __restrict__ u,
                                    __bf16* __restrict__ out, size_t n) {
  size_t i = (size_t)blockIdx.x * blockDim.x + threadIdx.x;
  if (i < n) {
    float gv = g[i];
    out[i] = (__bf16)((gv / (1.f + expf(-gv))) * u[i]);
  }
}

// ---------------- 9) gate blend + scatter back ----------------
__global__ void mod_blend_scatter_kernel(const float* __restrict__ x2,
                                         const float* __restrict__ x,
                                         const float* __restrict__ gate,
                                         const int* __restrict__ idx,
                                         float* __restrict__ out) {
  const int s = blockIdx.x;
  const int b = s / KK;
  const int tpos = idx[s];
  const float gt = gate[s];
  float* orow = out + ((size_t)b * TT + tpos) * DD;
  for (int c = threadIdx.x; c < DD; c += 256)
    orow[c] = gt * x2[(size_t)s * DD + c] + (1.f - gt) * x[(size_t)s * DD + c];
}

// ---------------- launch ----------------
extern "C" void kernel_launch(void* const* d_in, const int* in_sizes, int n_in,
                              void* d_out, int out_size, void* d_ws, size_t ws_size,
                              hipStream_t stream) {
  (void)in_sizes; (void)n_in; (void)out_size; (void)ws_size;
  const float* hidden   = (const float*)d_in[0];
  const float* router_w = (const float*)d_in[1];
  const float* router_b = (const float*)d_in[2];
  const float* ln1_w    = (const float*)d_in[3];
  const float* ln2_w    = (const float*)d_in[4];
  const float* wq       = (const float*)d_in[5];
  const float* bq       = (const float*)d_in[6];
  const float* wk       = (const float*)d_in[7];
  const float* bk       = (const float*)d_in[8];
  const float* wv       = (const float*)d_in[9];
  const float* bv       = (const float*)d_in[10];
  const float* wo       = (const float*)d_in[11];
  const float* w_gate   = (const float*)d_in[12];
  const float* w_up     = (const float*)d_in[13];
  const float* w_down   = (const float*)d_in[14];
  float* out = (float*)d_out;

  uint8_t* base = (uint8_t*)d_ws;
  size_t off = 0;
  auto alloc = [&](size_t bytes) -> void* {
    void* r = base + off;
    off += (bytes + 255) & ~(size_t)255;
    return r;
  };
  const size_t SD = (size_t)SS * DD, SI = (size_t)SS * II;
  float*  scores = (float*)alloc((size_t)BB * TT * 4);
  int*    idxb   = (int*)  alloc((size_t)SS * 4);
  float*  gateb  = (float*)alloc((size_t)SS * 4);
  __bf16* wqt    = (__bf16*)alloc((size_t)DD * DD * 2);
  __bf16* wkt    = (__bf16*)alloc((size_t)DD * DD * 2);
  __bf16* wvt    = (__bf16*)alloc((size_t)DD * DD * 2);
  __bf16* wot    = (__bf16*)alloc((size_t)DD * DD * 2);
  __bf16* wgt    = (__bf16*)alloc((size_t)II * DD * 2);
  __bf16* wut    = (__bf16*)alloc((size_t)II * DD * 2);
  __bf16* wdt    = (__bf16*)alloc((size_t)DD * II * 2);
  float*  xbuf   = (float*)alloc(SD * 4);
  __bf16* hb     = (__bf16*)alloc(SD * 2);
  float*  cosb   = (float*)alloc((size_t)SS * HDIM * 4);
  float*  sinb   = (float*)alloc((size_t)SS * HDIM * 4);
  float*  qf     = (float*)alloc(SD * 4);
  float*  kf     = (float*)alloc(SD * 4);
  float*  vf     = (float*)alloc(SD * 4);
  __bf16* qhb    = (__bf16*)alloc(SD * 2);
  __bf16* khb    = (__bf16*)alloc(SD * 2);
  __bf16* vtb    = (__bf16*)alloc(SD * 2);
  __bf16* attnb  = (__bf16*)alloc(SD * 2);
  float*  x1     = (float*)alloc(SD * 4);
  __bf16* h2b    = (__bf16*)alloc(SD * 2);
  float*  gbuf   = (float*)alloc(SI * 4);
  float*  ubuf   = (float*)alloc(SI * 4);
  __bf16* actb   = (__bf16*)alloc(SI * 2);
  float*  x2     = (float*)alloc(SD * 4);

  // output starts as a copy of hidden_states (non-selected tokens pass through)
  (void)hipMemcpyAsync(out, hidden, (size_t)BB * TT * DD * sizeof(float),
                       hipMemcpyDeviceToDevice, stream);

  mod_router_kernel<<<(BB * TT) / 8, 256, 0, stream>>>(hidden, router_w, router_b, scores);
  mod_topk_kernel<<<BB, 1024, 0, stream>>>(scores, idxb, gateb);

  dim3 tb(32, 8);
  mod_transpose_cast_kernel<<<dim3(DD/32, DD/32), tb, 0, stream>>>(wq, wqt, DD, DD);
  mod_transpose_cast_kernel<<<dim3(DD/32, DD/32), tb, 0, stream>>>(wk, wkt, DD, DD);
  mod_transpose_cast_kernel<<<dim3(DD/32, DD/32), tb, 0, stream>>>(wv, wvt, DD, DD);
  mod_transpose_cast_kernel<<<dim3(DD/32, DD/32), tb, 0, stream>>>(wo, wot, DD, DD);
  mod_transpose_cast_kernel<<<dim3(II/32, DD/32), tb, 0, stream>>>(w_gate, wgt, DD, II);
  mod_transpose_cast_kernel<<<dim3(II/32, DD/32), tb, 0, stream>>>(w_up,   wut, DD, II);
  mod_transpose_cast_kernel<<<dim3(DD/32, II/32), tb, 0, stream>>>(w_down, wdt, II, DD);

  mod_gather_rms_kernel<<<SS, 256, 0, stream>>>(hidden, idxb, ln1_w, xbuf, hb, cosb, sinb);

  mod_gemm_kernel<<<dim3(DD/128, SS/128), 256, 0, stream>>>(hb, wqt, qf, bq, nullptr, SS, DD, DD);
  mod_gemm_kernel<<<dim3(DD/128, SS/128), 256, 0, stream>>>(hb, wkt, kf, bk, nullptr, SS, DD, DD);
  mod_gemm_kernel<<<dim3(DD/128, SS/128), 256, 0, stream>>>(hb, wvt, vf, bv, nullptr, SS, DD, DD);

  mod_rope_cast_kernel<<<dim3(HH, SS), HDIM, 0, stream>>>(qf, kf, vf, cosb, sinb, qhb, khb, vtb);

  mod_flash_kernel<<<dim3(SS/64, HH), 128, 0, stream>>>(qhb, khb, vtb, attnb);

  // x1 = x + attn @ wo
  mod_gemm_kernel<<<dim3(DD/128, SS/128), 256, 0, stream>>>(attnb, wot, x1, nullptr, xbuf, SS, DD, DD);

  mod_gather_rms_kernel<<<SS, 256, 0, stream>>>(x1, nullptr, ln2_w, nullptr, h2b, nullptr, nullptr);

  mod_gemm_kernel<<<dim3(II/128, SS/128), 256, 0, stream>>>(h2b, wgt, gbuf, nullptr, nullptr, SS, II, DD);
  mod_gemm_kernel<<<dim3(II/128, SS/128), 256, 0, stream>>>(h2b, wut, ubuf, nullptr, nullptr, SS, II, DD);

  mod_silu_mul_kernel<<<(int)((SI + 255) / 256), 256, 0, stream>>>(gbuf, ubuf, actb, SI);

  // x2 = x1 + act @ w_down
  mod_gemm_kernel<<<dim3(DD/128, SS/128), 256, 0, stream>>>(actb, wdt, x2, nullptr, x1, SS, DD, II);

  mod_blend_scatter_kernel<<<SS, 256, 0, stream>>>(x2, xbuf, gateb, idxb, out);
}
